// Model_17119739641983
// MI455X (gfx1250) — matrix operand order repaired and checked by reading, not compiled, not run on hardware
//
#include <hip/hip_runtime.h>
#include <math.h>

typedef __bf16 bf16;
typedef __bf16 v16bf __attribute__((ext_vector_type(16)));
typedef float  v8f   __attribute__((ext_vector_type(8)));

union AB {
    v16bf v;
    float4 f4[2];
};

// Native fp32 -> bf16 conversion (lowers to v_cvt_*bf16_f32 on gfx1250,
// replacing the 3-VALU-per-element software RNE sequence).
__device__ __forceinline__ bf16 tobf(float f) { return (bf16)f; }

__device__ __forceinline__ v8f splat8(float f) {
    v8f v;
#pragma unroll
    for (int i = 0; i < 8; ++i) v[i] = f;
    return v;
}

// ---------------------------------------------------------------------------
// Generic GEMM with fused residual: C[M,N] = A[M,K]*B[K,N] (+ resid[M,N]).
// fp32 global, bf16 LDS tiles, v_wmma_f32_16x16x32_bf16, f32 accumulate.
// Block tile 64x128, BK=64, 8 waves (wave32), each wave owns 32x32 of C.
// Requires M%64==0, N%128==0, K%64==0 (true for every GEMM in this model).
// ---------------------------------------------------------------------------
__global__ __launch_bounds__(256) void gemm_bf16_wmma(
    const float* __restrict__ Al, const float* __restrict__ Bl,
    float* __restrict__ C, const float* __restrict__ resid,
    int M, int N, int K)
{
    __shared__ bf16 As[64][64];    // [m][k]  8 KB
    __shared__ bf16 Bs[128][64];   // [n][k] 16 KB (transposed: K contiguous)

    const int tid  = threadIdx.x;
    const int lane = tid & 31;
    const int wave = tid >> 5;
    const int wm   = wave >> 2;        // 0..1
    const int wn   = wave & 3;         // 0..3
    const int bm   = blockIdx.y * 64;
    const int bn   = blockIdx.x * 128;

    v8f acc[2][2];
#pragma unroll
    for (int i = 0; i < 2; ++i)
#pragma unroll
        for (int j = 0; j < 2; ++j) acc[i][j] = splat8(0.f);

    const int ar  = tid >> 2;          // 0..63 : A row
    const int ac0 = (tid & 3) * 16;    // A col group

    for (int k0 = 0; k0 < K; k0 += 64) {
        // ---- stage A tile (64x64): 16 elems/thread --------------------------
        {
            const float* src = Al + (size_t)(bm + ar) * K + k0 + ac0;
#pragma unroll
            for (int i = 0; i < 16; i += 4) {
                float4 f = *(const float4*)(src + i);
                As[ar][ac0 + i + 0] = tobf(f.x);
                As[ar][ac0 + i + 1] = tobf(f.y);
                As[ar][ac0 + i + 2] = tobf(f.z);
                As[ar][ac0 + i + 3] = tobf(f.w);
            }
            if (k0 + 64 < K)
                __builtin_prefetch(src + 64, 0, 3);   // global_prefetch_b8
        }
        // ---- stage B tile (64x128 -> Bs[n][k]): 32 elems/thread -------------
#pragma unroll
        for (int cc = 0; cc < 2; ++cc) {
            int c  = tid + cc * 256;        // 0..511
            int kk = c >> 3;                // 0..63
            int n0 = (c & 7) * 16;          // 0..112
            const float* src = Bl + (size_t)(k0 + kk) * N + bn + n0;
#pragma unroll
            for (int i = 0; i < 16; i += 4) {
                float4 f = *(const float4*)(src + i);
                Bs[n0 + i + 0][kk] = tobf(f.x);
                Bs[n0 + i + 1][kk] = tobf(f.y);
                Bs[n0 + i + 2][kk] = tobf(f.z);
                Bs[n0 + i + 3][kk] = tobf(f.w);
            }
            if (k0 + 64 < K)
                __builtin_prefetch(src + (size_t)64 * N, 0, 3);
        }
        __syncthreads();

        // ---- compute: 2 K-steps x 4 WMMAs -----------------------------------
#pragma unroll
        for (int ks = 0; ks < 64; ks += 32) {
            AB a[2], b[2];
            const int am = wm * 32 + (lane & 15);
            const int ak = ks + ((lane < 16) ? 0 : 8);
            a[0].f4[0] = *(const float4*)&As[am][ak];
            a[0].f4[1] = *(const float4*)&As[am][ak + 16];
            a[1].f4[0] = *(const float4*)&As[am + 16][ak];
            a[1].f4[1] = *(const float4*)&As[am + 16][ak + 16];
            const int bc = wn * 32 + (lane & 15);
            const int bk = ks + ((lane < 16) ? 0 : 8);
            b[0].f4[0] = *(const float4*)&Bs[bc][bk];
            b[0].f4[1] = *(const float4*)&Bs[bc][bk + 16];
            b[1].f4[0] = *(const float4*)&Bs[bc + 16][bk];
            b[1].f4[1] = *(const float4*)&Bs[bc + 16][bk + 16];
#pragma unroll
            for (int i = 0; i < 2; ++i)
#pragma unroll
                for (int j = 0; j < 2; ++j)
                    acc[i][j] = __builtin_amdgcn_wmma_f32_16x16x32_bf16(
                        false, a[i].v, false, b[j].v, (short)0, acc[i][j],
                        false, false);
        }
        __syncthreads();
    }

    // ---- writeback (C layout: row = vgpr + 8*(lane>=16), col = lane&15) -----
#pragma unroll
    for (int i = 0; i < 2; ++i)
#pragma unroll
        for (int j = 0; j < 2; ++j)
#pragma unroll
            for (int r = 0; r < 8; ++r) {
                int row = bm + wm * 32 + i * 16 + r + ((lane < 16) ? 0 : 8);
                int col = bn + wn * 32 + j * 16 + (lane & 15);
                size_t idx = (size_t)row * N + col;
                float val = acc[i][j][r];
                if (resid) val += resid[idx];
                C[idx] = val;
            }
}

// ---------------------------------------------------------------------------
// Flash attention (causal, GQA rep=4). One wave per 16-row Q tile, 4 waves
// per block. WMMA for QK^T and P.V, online softmax via __shfl_xor within
// 16-lane halves (matches C-layout row striping exactly).
// q: [2048, 32*64], k/v: [2048, 8*64], sa: [2048, 2048].
// ---------------------------------------------------------------------------
__global__ __launch_bounds__(128) void attn_kernel(
    const float* __restrict__ q, const float* __restrict__ k,
    const float* __restrict__ v, float* __restrict__ sa)
{
    __shared__ bf16 Qt[4][16][64];
    __shared__ bf16 Kt[4][32][64];
    __shared__ bf16 Vt[4][64][32];   // [dh][key] (K contiguous for B frags)
    __shared__ bf16 Pt[4][16][32];

    const int lane = threadIdx.x & 31;
    const int w    = threadIdx.x >> 5;
    const int head = blockIdx.y;
    const int kvh  = head >> 2;
    const int q0   = (blockIdx.x * 4 + w) * 16;

    // stage Q tile
    for (int idx = lane; idx < 16 * 64; idx += 32) {
        int r = idx >> 6, c = idx & 63;
        Qt[w][r][c] = tobf(q[(size_t)(q0 + r) * 2048 + head * 64 + c]);
    }
    asm volatile("s_wait_dscnt 0" ::: "memory");

    AB aq[2];
#pragma unroll
    for (int ks = 0; ks < 2; ++ks) {
        int am = lane & 15;
        int ak = ks * 32 + ((lane < 16) ? 0 : 8);
        aq[ks].f4[0] = *(const float4*)&Qt[w][am][ak];
        aq[ks].f4[1] = *(const float4*)&Qt[w][am][ak + 16];
    }

    v8f o[4];
#pragma unroll
    for (int j = 0; j < 4; ++j) o[j] = splat8(0.f);
    v8f m_i = splat8(-1e30f);
    v8f l_i = splat8(0.f);

    const int kend = q0 + 16;
    for (int k0 = 0; k0 < kend; k0 += 32) {
        // stage K (row-major) and V (transposed) tiles
        for (int idx = lane; idx < 32 * 64; idx += 32) {
            int kk = idx >> 6, d = idx & 63;
            size_t g = (size_t)(k0 + kk) * 512 + kvh * 64 + d;
            Kt[w][kk][d] = tobf(k[g]);
            Vt[w][d][kk] = tobf(v[g]);
        }
        asm volatile("s_wait_dscnt 0" ::: "memory");

        // S = Q K^T * scale  (two 16x16 C tiles over 32 keys)
        v8f s[2];
#pragma unroll
        for (int kt = 0; kt < 2; ++kt) {
            v8f sc = splat8(0.f);
#pragma unroll
            for (int ks = 0; ks < 2; ++ks) {
                AB bk;
                int key = kt * 16 + (lane & 15);
                int d0  = ks * 32 + ((lane < 16) ? 0 : 8);
                bk.f4[0] = *(const float4*)&Kt[w][key][d0];
                bk.f4[1] = *(const float4*)&Kt[w][key][d0 + 16];
                sc = __builtin_amdgcn_wmma_f32_16x16x32_bf16(
                    false, aq[ks].v, false, bk.v, (short)0, sc, false, false);
            }
            s[kt] = sc * 0.125f;   // 1/sqrt(64)
        }
        // causal mask
#pragma unroll
        for (int kt = 0; kt < 2; ++kt) {
            int key   = k0 + kt * 16 + (lane & 15);
            int rbase = q0 + ((lane < 16) ? 0 : 8);
#pragma unroll
            for (int r = 0; r < 8; ++r)
                if (key > rbase + r) s[kt][r] = -1e30f;
        }
        // row max (across 32 keys): tile-pair max then 16-lane-half reduction
        v8f rowm;
#pragma unroll
        for (int r = 0; r < 8; ++r) rowm[r] = fmaxf(s[0][r], s[1][r]);
#pragma unroll
        for (int msk = 1; msk <= 8; msk <<= 1)
#pragma unroll
            for (int r = 0; r < 8; ++r)
                rowm[r] = fmaxf(rowm[r], __shfl_xor(rowm[r], msk, 32));

        v8f m_new, alpha;
#pragma unroll
        for (int r = 0; r < 8; ++r) {
            m_new[r] = fmaxf(m_i[r], rowm[r]);
            alpha[r] = __expf(m_i[r] - m_new[r]);
        }
        // P = exp(S - m_new), row sums
        v8f rs = splat8(0.f);
#pragma unroll
        for (int kt = 0; kt < 2; ++kt)
#pragma unroll
            for (int r = 0; r < 8; ++r) {
                float p = __expf(s[kt][r] - m_new[r]);
                s[kt][r] = p;
                rs[r] += p;
            }
#pragma unroll
        for (int msk = 1; msk <= 8; msk <<= 1)
#pragma unroll
            for (int r = 0; r < 8; ++r)
                rs[r] += __shfl_xor(rs[r], msk, 32);
#pragma unroll
        for (int r = 0; r < 8; ++r) l_i[r] = l_i[r] * alpha[r] + rs[r];
#pragma unroll
        for (int j = 0; j < 4; ++j)
#pragma unroll
            for (int r = 0; r < 8; ++r) o[j][r] *= alpha[r];

        // P: C-layout -> A-layout through LDS
#pragma unroll
        for (int kt = 0; kt < 2; ++kt)
#pragma unroll
            for (int r = 0; r < 8; ++r) {
                int mrow = r + ((lane < 16) ? 0 : 8);
                Pt[w][mrow][kt * 16 + (lane & 15)] = tobf(s[kt][r]);
            }
        asm volatile("s_wait_dscnt 0" ::: "memory");

        AB ap;
        {
            int row = lane & 15;
            int pk  = (lane < 16) ? 0 : 8;
            ap.f4[0] = *(const float4*)&Pt[w][row][pk];
            ap.f4[1] = *(const float4*)&Pt[w][row][pk + 16];
        }
        // O += P V
#pragma unroll
        for (int j = 0; j < 4; ++j) {
            AB bv;
            int d    = j * 16 + (lane & 15);
            int key0 = (lane < 16) ? 0 : 8;
            bv.f4[0] = *(const float4*)&Vt[w][d][key0];
            bv.f4[1] = *(const float4*)&Vt[w][d][key0 + 16];
            o[j] = __builtin_amdgcn_wmma_f32_16x16x32_bf16(
                false, ap.v, false, bv.v, (short)0, o[j], false, false);
        }
        m_i = m_new;
    }

    v8f linv;
#pragma unroll
    for (int r = 0; r < 8; ++r) linv[r] = 1.f / l_i[r];
#pragma unroll
    for (int j = 0; j < 4; ++j)
#pragma unroll
        for (int r = 0; r < 8; ++r) {
            int row = q0 + r + ((lane < 16) ? 0 : 8);
            int col = head * 64 + j * 16 + (lane & 15);
            sa[(size_t)row * 2048 + col] = o[j][r] * linv[r];
        }
}

// ---------------------------------------------------------------------------
// Elementwise / norm / rope / embed kernels
// ---------------------------------------------------------------------------
__global__ __launch_bounds__(256) void rmsnorm_kernel(
    const float* __restrict__ x, const float* __restrict__ w,
    float* __restrict__ y, int D_)
{
    __shared__ float red[256];
    const int row = blockIdx.x;
    const float* xr = x + (size_t)row * D_;
    float s = 0.f;
    for (int d = threadIdx.x; d < D_; d += 256) { float t = xr[d]; s += t * t; }
    red[threadIdx.x] = s;
    __syncthreads();
    for (int off = 128; off > 0; off >>= 1) {
        if (threadIdx.x < off) red[threadIdx.x] += red[threadIdx.x + off];
        __syncthreads();
    }
    float rms = rsqrtf(red[0] / (float)D_ + 1e-5f);
    float* yr = y + (size_t)row * D_;
    for (int d = threadIdx.x; d < D_; d += 256) yr[d] = xr[d] * rms * w[d];
}

__global__ __launch_bounds__(256) void rope_kernel(
    float* __restrict__ x, const float* __restrict__ cosT,
    const float* __restrict__ sinT, int nheads, int stride)
{
    int id = blockIdx.x * 256 + threadIdx.x;
    int total = 2048 * nheads * 32;
    if (id >= total) return;
    int p  = id & 31;
    int hh = (id >> 5) % nheads;
    int l  = id / (nheads * 32);
    float c = cosT[l * 32 + p], sn = sinT[l * 32 + p];
    size_t base = (size_t)l * stride + hh * 64 + 2 * p;
    float xr = x[base], xi = x[base + 1];
    x[base]     = xr * c - xi * sn;
    x[base + 1] = xr * sn + xi * c;
}

__global__ __launch_bounds__(256) void embed_kernel(
    const int* __restrict__ ti, const float* __restrict__ emb,
    float* __restrict__ x)
{
    int id = blockIdx.x * 256 + threadIdx.x;   // over 2048*2048
    int l = id >> 11, d = id & 2047;
    x[id] = emb[(size_t)ti[l] * 2048 + d];
}

__global__ __launch_bounds__(256) void silu_mul_kernel(
    float* __restrict__ g, const float* __restrict__ u, int n)
{
    int id = blockIdx.x * 256 + threadIdx.x;
    if (id < n) {
        float gv = g[id];
        g[id] = (gv / (1.f + __expf(-gv))) * u[id];
    }
}

// ---------------------------------------------------------------------------
static inline void launch_gemm(const float* A, const float* B, float* C,
                               const float* resid, int M, int N, int K,
                               hipStream_t s)
{
    dim3 grid(N / 128, M / 64);
    gemm_bf16_wmma<<<grid, 256, 0, s>>>(A, B, C, resid, M, N, K);
}

extern "C" void kernel_launch(void* const* d_in, const int* in_sizes, int n_in,
                              void* d_out, int out_size, void* d_ws, size_t ws_size,
                              hipStream_t stream)
{
    (void)in_sizes; (void)n_in; (void)out_size; (void)ws_size;
    const int*   ti    = (const int*)  d_in[0];
    const float* emb   = (const float*)d_in[1];
    const float* wq    = (const float*)d_in[2];
    const float* wk    = (const float*)d_in[3];
    const float* wv    = (const float*)d_in[4];
    const float* wa    = (const float*)d_in[5];
    const float* w_an  = (const float*)d_in[6];
    const float* w_fn  = (const float*)d_in[7];
    const float* wg    = (const float*)d_in[8];
    const float* wu    = (const float*)d_in[9];
    const float* wd    = (const float*)d_in[10];
    const float* w_on  = (const float*)d_in[11];
    const float* w_out = (const float*)d_in[12];
    const float* cosT  = (const float*)d_in[13];
    const float* sinT  = (const float*)d_in[14];
    float* out = (float*)d_out;

    const int L = 2048, D = 2048, Dh = 5632, V = 32000;
    const size_t LD  = (size_t)L * D;
    const size_t LK  = (size_t)L * 512;
    const size_t LDh = (size_t)L * Dh;

    float* ws  = (float*)d_ws;
    float* x   = ws;
    float* h   = ws + LD;          // normed activations
    float* qb  = ws + 2 * LD;
    float* sab = ws + 3 * LD;      // attention out
    float* kb  = ws + 4 * LD;
    float* vb  = ws + 4 * LD + LK;
    float* gb  = ws + 4 * LD + 2 * LK;
    float* ub  = gb + LDh;

    embed_kernel<<<(int)(LD / 256), 256, 0, stream>>>(ti, emb, x);

    for (int l = 0; l < 2; ++l) {
        const float* wql = wq + (size_t)l * D * D;
        const float* wkl = wk + (size_t)l * D * 512;
        const float* wvl = wv + (size_t)l * D * 512;
        const float* wal = wa + (size_t)l * D * D;
        const float* wgl = wg + (size_t)l * D * Dh;
        const float* wul = wu + (size_t)l * D * Dh;
        const float* wdl = wd + (size_t)l * Dh * D;

        rmsnorm_kernel<<<L, 256, 0, stream>>>(x, w_an + (size_t)l * D, h, D);
        launch_gemm(h, wql, qb, nullptr, L, D, D, stream);
        launch_gemm(h, wkl, kb, nullptr, L, 512, D, stream);
        launch_gemm(h, wvl, vb, nullptr, L, 512, D, stream);
        rope_kernel<<<(L * 32 * 32) / 256, 256, 0, stream>>>(qb, cosT, sinT, 32, D);
        rope_kernel<<<(L * 8 * 32) / 256, 256, 0, stream>>>(kb, cosT, sinT, 8, 512);
        attn_kernel<<<dim3(L / 64, 32), 128, 0, stream>>>(qb, kb, vb, sab);
        launch_gemm(sab, wal, x, x, L, D, D, stream);   // x += sa @ wa

        rmsnorm_kernel<<<L, 256, 0, stream>>>(x, w_fn + (size_t)l * D, h, D);
        launch_gemm(h, wgl, gb, nullptr, L, Dh, D, stream);
        launch_gemm(h, wul, ub, nullptr, L, Dh, D, stream);
        silu_mul_kernel<<<(int)(LDh / 256), 256, 0, stream>>>(gb, ub, (int)LDh);
        launch_gemm(gb, wdl, x, x, L, D, Dh, stream);   // x += gu @ wd
    }

    rmsnorm_kernel<<<L, 256, 0, stream>>>(x, w_on, h, D);
    launch_gemm(h, w_out, out, nullptr, L, V, D, stream);
}
